// DefConv_49005576848085
// MI455X (gfx1250) — matrix operand order repaired and checked by reading, compiled, not run on hardware
//
#include <hip/hip_runtime.h>
#include <hip/hip_bf16.h>
#include <stdint.h>

// ---------------- problem constants ----------------
#define BATCH 8
#define CIN   64
#define HW    96
#define PIX_PER_IMG (HW * HW)          // 9216
#define NPIX  (BATCH * PIX_PER_IMG)    // 73728
#define M576  576                      // C*K*K
#define TOPK  192                      // C*K
#define KDIM  384                      // 2*TOPK
#define OC    128
#define KBLKS (KDIM / 32)              // 12 WMMA K-steps

typedef __attribute__((ext_vector_type(16))) _Float16 v16h;
typedef __attribute__((ext_vector_type(8)))  float    v8f;

// ---------------------------------------------------------------------------
// Kernel 0: repack w_conv (OC x 384, f32) into a fragment-ready f16 layout.
// For WMMA B (32x16 f16), k order as A: lane (half,n) gets pairs (K,K+1), K = half*16+2v,
// v=0..7.  Store wtp[kblk][oc][pi][2] with pi = half*8+v so each lane's
// fragment is 8 consecutive dwords (32B, 32B-aligned) -> 2x global_load_b128.
// ---------------------------------------------------------------------------
__global__ __launch_bounds__(256) void wconv_pack_f16(const float* __restrict__ w,
                                                      _Float16* __restrict__ wtp) {
    int i = blockIdx.x * 256 + threadIdx.x;          // over KBLKS*OC*32
    if (i >= KBLKS * OC * 32) return;
    const int kblk = i / (OC * 32);
    const int rem  = i % (OC * 32);
    const int oc   = rem / 32;
    const int pi2  = rem % 32;                       // pair-slot*2 + element
    const int pi   = pi2 >> 1;                       // 0..15
    const int e    = pi2 & 1;
    const int half = pi >> 3;
    const int v    = pi & 7;
    const int k    = kblk * 32 + half * 16 + v * 2 + e;
    wtp[i] = (_Float16)w[oc * KDIM + k];
}

// ---------------------------------------------------------------------------
// Kernel 1 (fused): depthwise 3x3 r/k conv -> softmax(576) -> rank-select
// top-192 (sorted desc, stable) -> y (f16, [pix][384]).  One wave32/pixel,
// 8 waves per block.  x (18.9MB) + weights stay L2-resident so recompute
// beats materializing the 340MB r/k tensors to HBM.
// ---------------------------------------------------------------------------
__global__ __launch_bounds__(256) void fused_rk_softmax_topk(
        const float* __restrict__ x,
        const float* __restrict__ w_r,
        const float* __restrict__ w_k,
        _Float16* __restrict__ yh) {
    __shared__ float pbuf[8][M576];
    __shared__ float kbuf[8][M576];

    const int wave = threadIdx.x >> 5;
    const int lane = threadIdx.x & 31;
    const int pix  = blockIdx.x * 8 + wave;

    const int b  = pix / PIX_PER_IMG;
    const int hw = pix % PIX_PER_IMG;
    const int h  = hw / HW;
    const int w0 = hw % HW;

    const float* xb = x + (size_t)b * CIN * PIX_PER_IMG;

    // ---- phase 1: depthwise conv, 18 of the 576 channels per lane ----
    float rloc[18];
    #pragma unroll
    for (int t = 0; t < 18; ++t) {
        const int j = t * 32 + lane;              // unfolded channel 0..575
        const int c = j / 9;                      // input channel
        const float* wr = w_r + (size_t)j * 9;    // (576,1,3,3)
        const float* wk = w_k + (size_t)j * 9;
        const float* xc = xb + (size_t)c * PIX_PER_IMG;
        float rv = 0.f, kv = 0.f;
        #pragma unroll
        for (int ii = 0; ii < 3; ++ii) {
            const int hy = h - 1 + ii;
            if (hy < 0 || hy >= HW) continue;
            #pragma unroll
            for (int jj = 0; jj < 3; ++jj) {
                const int wx = w0 - 1 + jj;
                if (wx < 0 || wx >= HW) continue;
                const float xv = xc[hy * HW + wx];
                rv += xv * wr[ii * 3 + jj];
                kv += xv * wk[ii * 3 + jj];
            }
        }
        rloc[t] = rv;
        kbuf[wave][j] = kv;
    }

    // ---- phase 2: softmax over 576 (wave32 reductions) ----
    float mx = rloc[0];
    #pragma unroll
    for (int t = 1; t < 18; ++t) mx = fmaxf(mx, rloc[t]);
    #pragma unroll
    for (int o = 16; o > 0; o >>= 1) mx = fmaxf(mx, __shfl_xor(mx, o, 32));

    float pj[18];
    float s = 0.f;
    #pragma unroll
    for (int t = 0; t < 18; ++t) { pj[t] = __expf(rloc[t] - mx); s += pj[t]; }
    #pragma unroll
    for (int o = 16; o > 0; o >>= 1) s += __shfl_xor(s, o, 32);
    const float inv = 1.0f / s;

    #pragma unroll
    for (int t = 0; t < 18; ++t) {
        pj[t] *= inv;                              // lane's probabilities
        pbuf[wave][t * 32 + lane] = pj[t];
    }

    __syncthreads();   // make per-wave LDS tiles visible for cross-lane scan

    // ---- phase 3: single LDS sweep computing all 18 ranks at once.
    // rank = #{>pj} + #{==pj with smaller idx} reproduces lax.top_k's
    // descending order with stable tie-break; rank<192 scatters straight
    // into the sorted output slot (no explicit sort). ----
    int cnt[18];
    #pragma unroll
    for (int t = 0; t < 18; ++t) cnt[t] = 0;

    const float4* p4 = (const float4*)pbuf[wave];  // rows are 16B-aligned
    for (int o4 = 0; o4 < M576 / 4; ++o4) {        // 144 x ds_load_b128
        const float4 po = p4[o4];
        const int ob = o4 * 4;
        #pragma unroll
        for (int t = 0; t < 18; ++t) {
            const int j = t * 32 + lane;
            const float pv = pj[t];
            cnt[t] += (po.x > pv) | ((po.x == pv) & (ob + 0 < j));
            cnt[t] += (po.y > pv) | ((po.y == pv) & (ob + 1 < j));
            cnt[t] += (po.z > pv) | ((po.z == pv) & (ob + 2 < j));
            cnt[t] += (po.w > pv) | ((po.w == pv) & (ob + 3 < j));
        }
    }

    const size_t ybase = (size_t)pix * KDIM;
    #pragma unroll
    for (int t = 0; t < 18; ++t) {
        if (cnt[t] < TOPK) {
            yh[ybase + cnt[t]]        = (_Float16)pj[t];                // topk_r
            yh[ybase + TOPK + cnt[t]] = (_Float16)kbuf[wave][t * 32 + lane]; // k
        }
    }
}

// ---------------------------------------------------------------------------
// Kernel 2: out[pix, oc] = sum_k y[pix,k] * w[oc,k] + bias[oc]
// 73728 x 128 x 384 GEMM via v_wmma_f32_16x16x32_f16.
// Block = 256 threads = 8 waves; block owns 16 pixels, wave w owns oc tile
// [16w, 16w+16).  A and B fragments each load as 2x global_load_b128.
// ---------------------------------------------------------------------------
__global__ __launch_bounds__(256) void gemm_y_w(
        const _Float16* __restrict__ yh,
        const _Float16* __restrict__ wtp,
        const float* __restrict__ bias,
        float* __restrict__ out) {
    const int wave = threadIdx.x >> 5;
    const int lane = threadIdx.x & 31;
    const int half = lane >> 4;        // K sub-block selector
    const int mn   = lane & 15;        // A row (pixel) / B col (oc)
    const int p0   = blockIdx.x * 16;  // pixel tile base
    const int n0   = wave * 16;        // oc tile base

    v8f acc = {};

    // A: 2 f16 per dword; lane's 8 dwords per fragment (per ISA 7.12.2).
    const uint32_t* arow32 =
        (const uint32_t*)(yh + (size_t)(p0 + mn) * KDIM);
    // B: fragment-ready pack; lane's 8 dwords are consecutive.
    const uint32_t* bbase =
        (const uint32_t*)wtp + ((size_t)(n0 + mn)) * 16 + half * 8;

    #pragma unroll
    for (int kb = 0; kb < KDIM; kb += 32) {
        v16h a;
        uint32_t* au = (uint32_t*)&a;
        #pragma unroll
        for (int v = 0; v < 8; ++v) {
            const int kd = half * 8 + v; // dword idx
            au[v] = arow32[(kb >> 1) + kd];
        }

        v16h bf;
        uint32_t* bu = (uint32_t*)&bf;
        const uint32_t* bptr = bbase + (size_t)(kb >> 5) * (OC * 16);
        #pragma unroll
        for (int v = 0; v < 8; ++v) bu[v] = bptr[v];

        __builtin_prefetch(arow32 + (kb >> 1) + 16, 0, 1);  // global_prefetch_b8

        acc = __builtin_amdgcn_wmma_f32_16x16x32_f16(
            /*neg_a=*/false, a, /*neg_b=*/false, bf,
            /*c_mod=*/(short)0, acc, /*reuse_a=*/false, /*reuse_b=*/false);
    }

    // C/D layout: VGPR i -> M = half*8 + i, N = lane&15
    const int oc = n0 + mn;
    const float bv = bias[oc];
    #pragma unroll
    for (int i = 0; i < 8; ++i) {
        const int pix = p0 + half * 8 + i;
        const int b   = pix / PIX_PER_IMG;
        const int hw  = pix % PIX_PER_IMG;
        out[((size_t)b * OC + oc) * PIX_PER_IMG + hw] = acc[i] + bv;
    }
}

// ---------------------------------------------------------------------------
extern "C" void kernel_launch(void* const* d_in, const int* in_sizes, int n_in,
                              void* d_out, int out_size, void* d_ws, size_t ws_size,
                              hipStream_t stream) {
    const float* x      = (const float*)d_in[0];
    const float* w_r    = (const float*)d_in[1];
    const float* w_k    = (const float*)d_in[2];
    const float* w_conv = (const float*)d_in[3];
    const float* b_conv = (const float*)d_in[4];
    float* out = (float*)d_out;

    char* ws = (char*)d_ws;
    _Float16* wtp = (_Float16*)ws;                       // 12*128*32 f16 (96 KB)
    _Float16* yh  = (_Float16*)(ws + 128 * 1024);        // NPIX*384 f16 (~54 MB)

    // 0) repack + f16-convert 1x1 conv weights into WMMA-fragment layout
    wconv_pack_f16<<<(KBLKS * OC * 32 + 255) / 256, 256, 0, stream>>>(w_conv, wtp);

    // 1) fused dwconv + softmax + top-k -> y  (8 pixels / block)
    fused_rk_softmax_topk<<<NPIX / 8, 256, 0, stream>>>(x, w_r, w_k, yh);

    // 2) WMMA GEMM + bias  (16 pixels x 128 oc per block)
    gemm_y_w<<<NPIX / 16, 256, 0, stream>>>(yh, wtp, b_conv, out);
}
